// LightGlue_77154792505740
// MI455X (gfx1250) — compile-verified, matrix operand order checked by reading, not run on hardware
//
#include <hip/hip_runtime.h>
#include <hip/hip_bf16.h>

#define HH 4
#define MM 1536
#define NNP 1536
#define CC 256
#define DD 64
#define LLAYERS 9

typedef __attribute__((ext_vector_type(16))) __bf16 v16bf;
typedef __attribute__((ext_vector_type(8)))  __bf16 v8bf;
typedef __attribute__((ext_vector_type(8)))  float  v8f;

__device__ inline __bf16 f2bf(float f) {
    unsigned u = __builtin_bit_cast(unsigned, f);
    unsigned r = u + 0x7FFFu + ((u >> 16) & 1u);   // round-to-nearest-even
    unsigned short h = (unsigned short)(r >> 16);
    return __builtin_bit_cast(__bf16, h);
}

__device__ inline float gelu_exact(float x) {
    return 0.5f * x * (1.0f + erff(x * 0.70710678118654752f));
}

__device__ inline float logsig(float x) {
    return fminf(x, 0.0f) - log1pf(expf(-fabsf(x)));
}

// ---------------------------------------------------------------------------
// Batched WMMA GEMM on pre-converted bf16 operands.
//   A: row-major (M,K) bf16.  B: (N,K) bf16 (i.e. already transposed).
//   One wave computes a 32x64 output block: 2 A-fragments, 4 B-fragments,
//   8 v_wmma per 32-wide K step (each B-fragment reused for both M-tiles).
//   All fragment loads are 16-byte b128.
//   OUT=0: f32 C (+bias+res+alpha).  OUT=1: bf16 C.  OUT=2: bf16 C transposed.
// ---------------------------------------------------------------------------
struct GemmP {
    const __bf16* A; const __bf16* B; void* C;
    const float* bias; const float* res;
    long long sA, sB, sC, sbias, sres;
    int lda, ldb, ldc, ldres;
    int M, N, K;
    float alpha;
};

template <int OUT>
__global__ __launch_bounds__(32) void gemm_bf16(GemmP p) {
    const int lane = threadIdx.x & 31;
    const int r    = lane & 15;
    const int hi   = (lane >> 4) & 1;
    const int tn0  = blockIdx.x * 64;
    const int tm   = blockIdx.y * 32;
    const int b    = blockIdx.z;

    const __bf16* A = p.A + (long long)b * p.sA;
    const __bf16* B = p.B + (long long)b * p.sB;
    const int kbase = hi * 8;

    const __bf16* Arow0 = A + (long long)(tm + r) * p.lda + kbase;
    const __bf16* Arow1 = A + (long long)(tm + 16 + r) * p.lda + kbase;
    const __bf16* Brow[4];
#pragma unroll
    for (int t = 0; t < 4; ++t)
        Brow[t] = B + (long long)(tn0 + t * 16 + r) * p.ldb + kbase;

    v8f acc0[4] = {};
    v8f acc1[4] = {};
    for (int k0 = 0; k0 < p.K; k0 += 32) {
        v8bf a00 = *(const v8bf*)(Arow0 + k0);
        v8bf a01 = *(const v8bf*)(Arow0 + k0 + 16);
        v8bf a10 = *(const v8bf*)(Arow1 + k0);
        v8bf a11 = *(const v8bf*)(Arow1 + k0 + 16);
        v16bf af0 = __builtin_shufflevector(a00, a01, 0, 1, 2, 3, 4, 5, 6, 7,
                                            8, 9, 10, 11, 12, 13, 14, 15);
        v16bf af1 = __builtin_shufflevector(a10, a11, 0, 1, 2, 3, 4, 5, 6, 7,
                                            8, 9, 10, 11, 12, 13, 14, 15);
#pragma unroll
        for (int t = 0; t < 4; ++t) {
            v8bf b0 = *(const v8bf*)(Brow[t] + k0);
            v8bf b1 = *(const v8bf*)(Brow[t] + k0 + 16);
            v16bf bfv = __builtin_shufflevector(b0, b1, 0, 1, 2, 3, 4, 5, 6, 7,
                                                8, 9, 10, 11, 12, 13, 14, 15);
            acc0[t] = __builtin_amdgcn_wmma_f32_16x16x32_bf16(
                          false, af0, false, bfv, (short)0, acc0[t], false, false);
            acc1[t] = __builtin_amdgcn_wmma_f32_16x16x32_bf16(
                          false, af1, false, bfv, (short)0, acc1[t], false, false);
        }
    }

    const float* bias = p.bias ? (p.bias + (long long)b * p.sbias) : nullptr;
    const float* res  = p.res  ? (p.res  + (long long)b * p.sres)  : nullptr;
#pragma unroll
    for (int half = 0; half < 2; ++half) {
        v8f* acc = half ? acc1 : acc0;
        const int rbase = tm + half * 16 + hi * 8;
#pragma unroll
        for (int t = 0; t < 4; ++t) {
            const int col = tn0 + t * 16 + r;
#pragma unroll
            for (int e = 0; e < 8; ++e) {
                const int row = rbase + e;
                float v = acc[t][e] * p.alpha;
                if (bias) v += bias[col];
                if (res)  v += res[(long long)row * p.ldres + col];
                if (OUT == 0)
                    ((float*)p.C)[(long long)b * p.sC + (long long)row * p.ldc + col] = v;
                else if (OUT == 1)
                    ((__bf16*)p.C)[(long long)b * p.sC + (long long)row * p.ldc + col] = f2bf(v);
                else
                    ((__bf16*)p.C)[(long long)b * p.sC + (long long)col * p.ldc + row] = f2bf(v);
            }
        }
    }
}

// ---------------------------------------------------------------------------
// Conversion / elementwise kernels
// ---------------------------------------------------------------------------
__global__ void conv_bf16(const float* src, __bf16* dst, long long n) {
    long long i = (long long)blockIdx.x * blockDim.x + threadIdx.x;
    if (i < n) dst[i] = f2bf(src[i]);
}

__global__ void gelu_conv(const float* src, __bf16* dst, long long n) {
    long long i = (long long)blockIdx.x * blockDim.x + threadIdx.x;
    if (i < n) dst[i] = f2bf(gelu_exact(src[i]));
}

// Wt[l, n, k] = W[l, k, n]
__global__ void tconv(const float* Wsrc, __bf16* Wt, long long total, int K, int N) {
    long long i = (long long)blockIdx.x * blockDim.x + threadIdx.x;
    if (i >= total) return;
    long long kn = (long long)K * N;
    long long l = i / kn, rem = i % kn;
    int k = (int)(rem / N), n = (int)(rem % N);
    Wt[l * kn + (long long)n * K + k] = f2bf(Wsrc[i]);
}

// P[b,m,n] = exp(S - rowlse)
__global__ void prow_conv(const float* S, const float* lse, __bf16* P,
                          long long total, int Nc) {
    long long i = (long long)blockIdx.x * blockDim.x + threadIdx.x;
    if (i >= total) return;
    long long row = i / Nc;
    P[i] = f2bf(expf(S[i] - lse[row]));
}

// Pt[b,n,m] = exp(S[b,m,n] - collse[b,n])
__global__ void pcolT_conv(const float* S, const float* cl, __bf16* Pt,
                           int Mr, int Nc, int batch) {
    long long i = (long long)blockIdx.x * blockDim.x + threadIdx.x;
    long long mn = (long long)Mr * Nc;
    if (i >= (long long)batch * mn) return;
    long long b = i / mn, rem = i % mn;
    int m = (int)(rem / Nc), n = (int)(rem % Nc);
    Pt[b * mn + (long long)n * Mr + m] = f2bf(expf(S[i] - cl[b * Nc + n]));
}

// Keypoint normalization stats: o = {shift_x, shift_y, scale}
__global__ void norm_kpts_kernel(const float* k, int M, float* o) {
    __shared__ float s0[256], s1[256], s2[256], s3[256];
    int t = threadIdx.x;
    float mnx = 3.4e38f, mny = 3.4e38f, mxx = -3.4e38f, mxy = -3.4e38f;
    for (int i = t; i < M; i += 256) {
        float x = k[2 * i], y = k[2 * i + 1];
        mnx = fminf(mnx, x); mxx = fmaxf(mxx, x);
        mny = fminf(mny, y); mxy = fmaxf(mxy, y);
    }
    s0[t] = mnx; s1[t] = mxx; s2[t] = mny; s3[t] = mxy; __syncthreads();
    for (int s = 128; s > 0; s >>= 1) {
        if (t < s) {
            s0[t] = fminf(s0[t], s0[t + s]); s1[t] = fmaxf(s1[t], s1[t + s]);
            s2[t] = fminf(s2[t], s2[t + s]); s3[t] = fmaxf(s3[t], s3[t + s]);
        }
        __syncthreads();
    }
    if (t == 0) {
        float sx = 1.0f + s1[0] - s0[0];
        float sy = 1.0f + s3[0] - s2[0];
        o[0] = sx * 0.5f; o[1] = sy * 0.5f; o[2] = fmaxf(sx, sy) * 0.5f;
    }
}

// Rotary positional encodings: cos/sin arrays of (M, 64), values repeated x2
__global__ void posenc_kernel(const float* kpts, const float* nrm, const float* peW,
                              float* cosO, float* sinO, int M) {
    int idx = blockIdx.x * blockDim.x + threadIdx.x;
    if (idx >= M * 32) return;
    int m = idx >> 5, j = idx & 31;
    float xn = (kpts[2 * m]     - nrm[0]) / nrm[2];
    float yn = (kpts[2 * m + 1] - nrm[1]) / nrm[2];
    float proj = xn * peW[j] + yn * peW[32 + j];
    float c = cosf(proj), s = sinf(proj);
    cosO[m * 64 + 2 * j] = c; cosO[m * 64 + 2 * j + 1] = c;
    sinO[m * 64 + 2 * j] = s; sinO[m * 64 + 2 * j + 1] = s;
}

// Split interleaved qkv (M,768) -> bf16 Q,K row-major (H,M,64) with RoPE,
// and V transposed (H,64,M) so it can be a (N,K) WMMA B-operand directly.
__global__ void rope_extract_bf(const float* qkv, const float* cosE, const float* sinE,
                                __bf16* Qr, __bf16* Kr, __bf16* Vt, int M) {
    int idx = blockIdx.x * blockDim.x + threadIdx.x;
    if (idx >= HH * M * 32) return;
    int i = idx & 31;
    int m = (idx >> 5) % M;
    int h = idx / (M * 32);
    const float* row = qkv + (long long)m * (3 * CC);
    int d0 = 2 * i, d1 = 2 * i + 1;
    float q0 = row[(h * DD + d0) * 3 + 0], q1 = row[(h * DD + d1) * 3 + 0];
    float k0 = row[(h * DD + d0) * 3 + 1], k1 = row[(h * DD + d1) * 3 + 1];
    float v0 = row[(h * DD + d0) * 3 + 2], v1 = row[(h * DD + d1) * 3 + 2];
    float c = cosE[m * 64 + d0], s = sinE[m * 64 + d0];
    long long o = ((long long)h * M + m) * DD;
    Qr[o + d0] = f2bf(q0 * c - q1 * s); Qr[o + d1] = f2bf(q1 * c + q0 * s);
    Kr[o + d0] = f2bf(k0 * c - k1 * s); Kr[o + d1] = f2bf(k1 * c + k0 * s);
    Vt[((long long)h * DD + d0) * M + m] = f2bf(v0);
    Vt[((long long)h * DD + d1) * M + m] = f2bf(v1);
}

// Row-wise log-sum-exp: one block per (flattened) row
__global__ void row_lse(const float* S, int Nc, float* lse) {
    __shared__ float red[256];
    long long row = blockIdx.x;
    const float* p = S + row * (long long)Nc;
    float mx = -3.4e38f;
    for (int i = threadIdx.x; i < Nc; i += 256) mx = fmaxf(mx, p[i]);
    red[threadIdx.x] = mx; __syncthreads();
    for (int s = 128; s > 0; s >>= 1) {
        if (threadIdx.x < s) red[threadIdx.x] = fmaxf(red[threadIdx.x], red[threadIdx.x + s]);
        __syncthreads();
    }
    mx = red[0]; __syncthreads();
    float sum = 0.0f;
    for (int i = threadIdx.x; i < Nc; i += 256) sum += expf(p[i] - mx);
    red[threadIdx.x] = sum; __syncthreads();
    for (int s = 128; s > 0; s >>= 1) {
        if (threadIdx.x < s) red[threadIdx.x] += red[threadIdx.x + s];
        __syncthreads();
    }
    if (threadIdx.x == 0) lse[row] = mx + logf(red[0]);
}

// Column-wise log-sum-exp: one block per (batch, col)
__global__ void col_lse(const float* S, int Mr, int Nc, float* lse) {
    __shared__ float red[256];
    long long b = blockIdx.x / Nc;
    int n = blockIdx.x % Nc;
    const float* p = S + b * (long long)Mr * Nc + n;
    float mx = -3.4e38f;
    for (int i = threadIdx.x; i < Mr; i += 256) mx = fmaxf(mx, p[(long long)i * Nc]);
    red[threadIdx.x] = mx; __syncthreads();
    for (int s = 128; s > 0; s >>= 1) {
        if (threadIdx.x < s) red[threadIdx.x] = fmaxf(red[threadIdx.x], red[threadIdx.x + s]);
        __syncthreads();
    }
    mx = red[0]; __syncthreads();
    float sum = 0.0f;
    for (int i = threadIdx.x; i < Mr; i += 256) sum += expf(p[(long long)i * Nc] - mx);
    red[threadIdx.x] = sum; __syncthreads();
    for (int s = 128; s > 0; s >>= 1) {
        if (threadIdx.x < s) red[threadIdx.x] += red[threadIdx.x + s];
        __syncthreads();
    }
    if (threadIdx.x == 0) lse[blockIdx.x] = mx + logf(red[0]);
}

// In-place LayerNorm over rows of length L
__global__ void ln_rows(float* X, const float* g, const float* b, int L) {
    __shared__ float rs[256], rq[256];
    long long row = blockIdx.x;
    float* p = X + row * (long long)L;
    float s = 0.0f, q = 0.0f;
    for (int i = threadIdx.x; i < L; i += 256) { float v = p[i]; s += v; q += v * v; }
    rs[threadIdx.x] = s; rq[threadIdx.x] = q; __syncthreads();
    for (int st = 128; st > 0; st >>= 1) {
        if (threadIdx.x < st) { rs[threadIdx.x] += rs[threadIdx.x + st];
                                rq[threadIdx.x] += rq[threadIdx.x + st]; }
        __syncthreads();
    }
    float mean = rs[0] / (float)L;
    float var  = rq[0] / (float)L - mean * mean;
    float inv  = rsqrtf(var + 1e-5f);
    for (int i = threadIdx.x; i < L; i += 256)
        p[i] = (p[i] - mean) * inv * g[i] + b[i];
}

__global__ void copy2d(const float* src, int lds, float* dst, int ldd, int rows, int cols) {
    long long idx = (long long)blockIdx.x * blockDim.x + threadIdx.x;
    if (idx >= (long long)rows * cols) return;
    int r = (int)(idx / cols), c = (int)(idx % cols);
    dst[(long long)r * ldd + c] = src[(long long)r * lds + c];
}

__global__ void zproj(const float* X, const float* zW, const float* zb, float* z, int M, int C) {
    int m = blockIdx.x * blockDim.x + threadIdx.x;
    if (m >= M) return;
    float s = zb[0];
    const float* row = X + (long long)m * C;
    for (int k = 0; k < C; ++k) s += row[k] * zW[k];
    z[m] = s;
}

__global__ void assemble_scores(const float* sim, const float* rl, const float* cl,
                                const float* z0, const float* z1, float* sc,
                                int Mr, int Nc) {
    long long idx = (long long)blockIdx.x * blockDim.x + threadIdx.x;
    long long tot = (long long)(Mr + 1) * (Nc + 1);
    if (idx >= tot) return;
    int m = (int)(idx / (Nc + 1)), n = (int)(idx % (Nc + 1));
    float v;
    if (m < Mr && n < Nc) {
        float s = sim[(long long)m * Nc + n];
        v = 2.0f * s - rl[m] - cl[n] + logsig(z0[m]) + logsig(z1[n]);
    } else if (m < Mr) v = logsig(-z0[m]);
    else if (n < Nc)   v = logsig(-z1[n]);
    else               v = 0.0f;
    sc[idx] = v;
}

__global__ void row_argmax(const float* sc, int ld, int Nc, float* maxv, int* arg) {
    __shared__ float rv[256]; __shared__ int ri[256];
    long long m = blockIdx.x;
    const float* p = sc + m * (long long)ld;
    float best = -3.4e38f; int bi = 0;
    for (int i = threadIdx.x; i < Nc; i += 256) {
        float v = p[i];
        if (v > best) { best = v; bi = i; }
    }
    rv[threadIdx.x] = best; ri[threadIdx.x] = bi; __syncthreads();
    for (int s = 128; s > 0; s >>= 1) {
        if (threadIdx.x < s && rv[threadIdx.x + s] > rv[threadIdx.x]) {
            rv[threadIdx.x] = rv[threadIdx.x + s];
            ri[threadIdx.x] = ri[threadIdx.x + s];
        }
        __syncthreads();
    }
    if (threadIdx.x == 0) { maxv[m] = rv[0]; arg[m] = ri[0]; }
}

__global__ void col_argmax(const float* sc, int ld, int Mr, int* arg) {
    __shared__ float rv[256]; __shared__ int ri[256];
    int n = blockIdx.x;
    float best = -3.4e38f; int bi = 0;
    for (int i = threadIdx.x; i < Mr; i += 256) {
        float v = sc[(long long)i * ld + n];
        if (v > best) { best = v; bi = i; }
    }
    rv[threadIdx.x] = best; ri[threadIdx.x] = bi; __syncthreads();
    for (int s = 128; s > 0; s >>= 1) {
        if (threadIdx.x < s && rv[threadIdx.x + s] > rv[threadIdx.x]) {
            rv[threadIdx.x] = rv[threadIdx.x + s];
            ri[threadIdx.x] = ri[threadIdx.x + s];
        }
        __syncthreads();
    }
    if (threadIdx.x == 0) arg[n] = ri[0];
}

__global__ void filt0(const int* m0r, const int* m1r, const float* max0,
                      float* m0o, float* ms0, int M, float th) {
    int i = blockIdx.x * blockDim.x + threadIdx.x;
    if (i >= M) return;
    int j = m0r[i];
    bool mutual = (m1r[j] == i);
    float s = mutual ? expf(max0[i]) : 0.0f;
    ms0[i] = s;
    bool valid = mutual && (s > th);
    m0o[i] = valid ? (float)j : -1.0f;
}

__global__ void filt1(const int* m0r, const int* m1r, const float* ms0,
                      const float* m0o, float* m1o, float* ms1, int N) {
    int j = blockIdx.x * blockDim.x + threadIdx.x;
    if (j >= N) return;
    int i = m1r[j];
    bool mutual = (m0r[i] == j);
    ms1[j] = mutual ? ms0[i] : 0.0f;
    bool valid = mutual && (m0o[i] >= 0.0f);
    m1o[j] = valid ? (float)i : -1.0f;
}

// ---------------------------------------------------------------------------
extern "C" void kernel_launch(void* const* d_in, const int* in_sizes, int n_in,
                              void* d_out, int out_size, void* d_ws, size_t ws_size,
                              hipStream_t stream) {
    (void)in_sizes; (void)n_in; (void)out_size; (void)ws_size;
    const float* kpts0 = (const float*)d_in[0];
    const float* kpts1 = (const float*)d_in[1];
    const float* desc0 = (const float*)d_in[2];
    const float* desc1 = (const float*)d_in[3];
    const float* peW   = (const float*)d_in[4];
    const float* sWqkv = (const float*)d_in[5];
    const float* sbqkv = (const float*)d_in[6];
    const float* sWout = (const float*)d_in[7];
    const float* sbout = (const float*)d_in[8];
    const float* sW1   = (const float*)d_in[9];
    const float* sb1   = (const float*)d_in[10];
    const float* sg    = (const float*)d_in[11];
    const float* sbeta = (const float*)d_in[12];
    const float* sW2   = (const float*)d_in[13];
    const float* sb2   = (const float*)d_in[14];
    const float* cWqk  = (const float*)d_in[15];
    const float* cbqk  = (const float*)d_in[16];
    const float* cWv   = (const float*)d_in[17];
    const float* cbv   = (const float*)d_in[18];
    const float* cWoutP= (const float*)d_in[19];
    const float* cboutP= (const float*)d_in[20];
    const float* cW1   = (const float*)d_in[21];
    const float* cb1   = (const float*)d_in[22];
    const float* cg    = (const float*)d_in[23];
    const float* cbeta = (const float*)d_in[24];
    const float* cW2   = (const float*)d_in[25];
    const float* cb2   = (const float*)d_in[26];
    const float* mW    = (const float*)d_in[27];
    const float* mb    = (const float*)d_in[28];
    const float* zW    = (const float*)d_in[29];
    const float* zb    = (const float*)d_in[30];

    char* wsb = (char*)d_ws;
    size_t off = 0;
    auto af32 = [&](size_t n) {
        off = (off + 63) & ~(size_t)63;
        float* p = (float*)(wsb + off); off += n * sizeof(float); return p;
    };
    auto abf = [&](size_t n) {
        off = (off + 63) & ~(size_t)63;
        __bf16* p = (__bf16*)(wsb + off); off += n * sizeof(__bf16); return p;
    };

    // fp32 scratch
    float* nrm0 = af32(8);
    float* nrm1 = af32(8);
    float* e0c = af32((size_t)MM * DD);
    float* e0s = af32((size_t)MM * DD);
    float* e1c = af32((size_t)NNP * DD);
    float* e1s = af32((size_t)NNP * DD);
    float* x0  = af32((size_t)MM * CC);
    float* x1  = af32((size_t)NNP * CC);
    float* qkv = af32((size_t)MM * 3 * CC);
    float* S   = af32((size_t)HH * MM * NNP);
    float* rl  = af32((size_t)HH * MM);
    float* cl  = af32((size_t)HH * NNP);
    float* cat = af32((size_t)MM * 2 * CC);
    float* hb  = af32((size_t)MM * 2 * CC);
    float* z0  = af32(MM);
    float* z1  = af32(NNP);
    float* max0 = af32(MM);
    int*   m0r  = (int*)af32(MM);
    int*   m1r  = (int*)af32(NNP);

    // bf16 scratch
    __bf16* x0bf = abf((size_t)MM * CC);
    __bf16* x1bf = abf((size_t)NNP * CC);
    __bf16* Qbf  = abf((size_t)HH * MM * DD);
    __bf16* Kbf  = abf((size_t)HH * MM * DD);
    __bf16* V0t  = abf((size_t)HH * DD * MM);
    __bf16* V1t  = abf((size_t)HH * DD * MM);
    __bf16* Pbf  = abf((size_t)HH * MM * NNP);
    __bf16* PtBf = abf((size_t)HH * NNP * MM);
    __bf16* ctxbf = abf((size_t)MM * CC);
    __bf16* catbf = abf((size_t)MM * 2 * CC);
    __bf16* hbg   = abf((size_t)MM * 2 * CC);
    __bf16* md0b  = abf((size_t)MM * CC);
    __bf16* md1b  = abf((size_t)NNP * CC);
    // transposed bf16 weights (N,K layout)
    __bf16* qkvT = abf((size_t)LLAYERS * 3 * CC * CC);
    __bf16* soutT = abf((size_t)LLAYERS * CC * CC);
    __bf16* s1T  = abf((size_t)LLAYERS * 2 * CC * 2 * CC);
    __bf16* s2T  = abf((size_t)LLAYERS * CC * 2 * CC);
    __bf16* cqkT = abf((size_t)LLAYERS * CC * CC);
    __bf16* cvT  = abf((size_t)LLAYERS * CC * CC);
    __bf16* coutT = abf((size_t)LLAYERS * CC * CC);
    __bf16* c1T  = abf((size_t)LLAYERS * 2 * CC * 2 * CC);
    __bf16* c2T  = abf((size_t)LLAYERS * CC * 2 * CC);
    __bf16* mWT  = abf((size_t)CC * CC);

    auto gemm = [&](int OUT, const __bf16* A, int lda, long long sA,
                    const __bf16* B, int ldb, long long sB,
                    void* C, int ldc, long long sC,
                    const float* bias, long long sbias,
                    const float* res, int ldres, long long sres,
                    int Mg, int Ng, int Kg, int batch, float alpha) {
        GemmP p;
        p.A = A; p.B = B; p.C = C; p.bias = bias; p.res = res;
        p.sA = sA; p.sB = sB; p.sC = sC; p.sbias = sbias; p.sres = sres;
        p.lda = lda; p.ldb = ldb; p.ldc = ldc; p.ldres = ldres;
        p.M = Mg; p.N = Ng; p.K = Kg; p.alpha = alpha;
        dim3 g((unsigned)(Ng / 64), (unsigned)(Mg / 32), (unsigned)batch);
        if (OUT == 0)      gemm_bf16<0><<<g, 32, 0, stream>>>(p);
        else if (OUT == 1) gemm_bf16<1><<<g, 32, 0, stream>>>(p);
        else               gemm_bf16<2><<<g, 32, 0, stream>>>(p);
    };
    auto grids = [](long long n) { return (unsigned)((n + 255) / 256); };

    // ---- one-time weight transpose+convert to bf16 (N,K) ----
    {
        long long t;
        t = (long long)LLAYERS * CC * 3 * CC;
        tconv<<<grids(t), 256, 0, stream>>>(sWqkv, qkvT, t, CC, 3 * CC);
        t = (long long)LLAYERS * CC * CC;
        tconv<<<grids(t), 256, 0, stream>>>(sWout, soutT, t, CC, CC);
        tconv<<<grids(t), 256, 0, stream>>>(cWqk, cqkT, t, CC, CC);
        tconv<<<grids(t), 256, 0, stream>>>(cWv,  cvT,  t, CC, CC);
        tconv<<<grids(t), 256, 0, stream>>>(cWoutP, coutT, t, CC, CC);
        t = (long long)LLAYERS * 2 * CC * 2 * CC;
        tconv<<<grids(t), 256, 0, stream>>>(sW1, s1T, t, 2 * CC, 2 * CC);
        tconv<<<grids(t), 256, 0, stream>>>(cW1, c1T, t, 2 * CC, 2 * CC);
        t = (long long)LLAYERS * 2 * CC * CC;
        tconv<<<grids(t), 256, 0, stream>>>(sW2, s2T, t, 2 * CC, CC);
        tconv<<<grids(t), 256, 0, stream>>>(cW2, c2T, t, 2 * CC, CC);
        t = (long long)CC * CC;
        tconv<<<grids(t), 256, 0, stream>>>(mW, mWT, t, CC, CC);
    }

    // ---- positional encodings ----
    norm_kpts_kernel<<<1, 256, 0, stream>>>(kpts0, MM, nrm0);
    norm_kpts_kernel<<<1, 256, 0, stream>>>(kpts1, NNP, nrm1);
    posenc_kernel<<<grids(MM * 32), 256, 0, stream>>>(kpts0, nrm0, peW, e0c, e0s, MM);
    posenc_kernel<<<grids(NNP * 32), 256, 0, stream>>>(kpts1, nrm1, peW, e1c, e1s, NNP);

    hipMemcpyAsync(x0, desc0, sizeof(float) * (size_t)MM * CC, hipMemcpyDeviceToDevice, stream);
    hipMemcpyAsync(x1, desc1, sizeof(float) * (size_t)NNP * CC, hipMemcpyDeviceToDevice, stream);
    conv_bf16<<<grids((long long)MM * CC), 256, 0, stream>>>(x0, x0bf, (long long)MM * CC);
    conv_bf16<<<grids((long long)NNP * CC), 256, 0, stream>>>(x1, x1bf, (long long)NNP * CC);

    const long long HMN = (long long)HH * MM * NNP;

    auto ffn = [&](float* x, __bf16* xbf, const __bf16* W1T, const float* b1,
                   const float* g_, const float* be, const __bf16* W2T, const float* b2) {
        copy2d<<<grids((long long)MM * CC), 256, 0, stream>>>(x, CC, cat, 2 * CC, MM, CC);
        conv_bf16<<<grids((long long)MM * 2 * CC), 256, 0, stream>>>(cat, catbf, (long long)MM * 2 * CC);
        gemm(0, catbf, 2 * CC, 0, W1T, 2 * CC, 0, hb, 2 * CC, 0,
             b1, 0, nullptr, 0, 0, MM, 2 * CC, 2 * CC, 1, 1.0f);
        ln_rows<<<MM, 256, 0, stream>>>(hb, g_, be, 2 * CC);
        gelu_conv<<<grids((long long)MM * 2 * CC), 256, 0, stream>>>(hb, hbg, (long long)MM * 2 * CC);
        gemm(0, hbg, 2 * CC, 0, W2T, 2 * CC, 0, x, CC, 0,
             b2, 0, x, CC, 0, MM, CC, 2 * CC, 1, 1.0f);
        conv_bf16<<<grids((long long)MM * CC), 256, 0, stream>>>(x, xbf, (long long)MM * CC);
    };

    auto selfblock = [&](float* x, __bf16* xbf, const float* ec, const float* es, int l) {
        gemm(0, xbf, CC, 0, qkvT + (size_t)l * 3 * CC * CC, CC, 0,
             qkv, 3 * CC, 0, sbqkv + (size_t)l * 3 * CC, 0, nullptr, 0, 0,
             MM, 3 * CC, CC, 1, 1.0f);
        rope_extract_bf<<<grids(HH * MM * 32), 256, 0, stream>>>(qkv, ec, es, Qbf, Kbf, V0t, MM);
        gemm(0, Qbf, DD, (long long)MM * DD, Kbf, DD, (long long)MM * DD,
             S, NNP, (long long)MM * NNP, nullptr, 0, nullptr, 0, 0,
             MM, NNP, DD, HH, 0.125f);
        row_lse<<<HH * MM, 256, 0, stream>>>(S, NNP, rl);
        prow_conv<<<grids(HMN), 256, 0, stream>>>(S, rl, Pbf, HMN, NNP);
        gemm(1, Pbf, NNP, (long long)MM * NNP, V0t, MM, (long long)DD * MM,
             ctxbf, CC, DD, nullptr, 0, nullptr, 0, 0,
             MM, DD, NNP, HH, 1.0f);
        gemm(0, ctxbf, CC, 0, soutT + (size_t)l * CC * CC, CC, 0,
             cat + CC, 2 * CC, 0, sbout + (size_t)l * CC, 0, nullptr, 0, 0,
             MM, CC, CC, 1, 1.0f);
        ffn(x, xbf, s1T + (size_t)l * 2 * CC * 2 * CC, sb1 + (size_t)l * 2 * CC,
            sg + (size_t)l * 2 * CC, sbeta + (size_t)l * 2 * CC,
            s2T + (size_t)l * CC * 2 * CC, sb2 + (size_t)l * CC);
    };

    auto crossblock = [&](int l) {
        const __bf16* WqkT = cqkT + (size_t)l * CC * CC;
        const __bf16* WvT  = cvT  + (size_t)l * CC * CC;
        const __bf16* WoT  = coutT + (size_t)l * CC * CC;
        const float* bqk = cbqk + (size_t)l * CC;
        const float* bv  = cbv  + (size_t)l * CC;
        const float* bo  = cboutP + (size_t)l * CC;
        // head-split projections (batch over heads = 64-row blocks of W^T)
        gemm(1, x0bf, CC, 0, WqkT, CC, (long long)DD * CC,
             Qbf, DD, (long long)MM * DD, bqk, DD, nullptr, 0, 0,
             MM, DD, CC, HH, 1.0f);
        gemm(1, x1bf, CC, 0, WqkT, CC, (long long)DD * CC,
             Kbf, DD, (long long)MM * DD, bqk, DD, nullptr, 0, 0,
             NNP, DD, CC, HH, 1.0f);
        gemm(2, x0bf, CC, 0, WvT, CC, (long long)DD * CC,
             V0t, MM, (long long)DD * MM, bv, DD, nullptr, 0, 0,
             MM, DD, CC, HH, 1.0f);
        gemm(2, x1bf, CC, 0, WvT, CC, (long long)DD * CC,
             V1t, NNP, (long long)DD * NNP, bv, DD, nullptr, 0, 0,
             NNP, DD, CC, HH, 1.0f);
        // sim = (qk0)(qk1)^T * D^-0.5
        gemm(0, Qbf, DD, (long long)MM * DD, Kbf, DD, (long long)NNP * DD,
             S, NNP, (long long)MM * NNP, nullptr, 0, nullptr, 0, 0,
             MM, NNP, DD, HH, 0.125f);
        row_lse<<<HH * MM, 256, 0, stream>>>(S, NNP, rl);
        col_lse<<<HH * NNP, 256, 0, stream>>>(S, MM, NNP, cl);
        prow_conv<<<grids(HMN), 256, 0, stream>>>(S, rl, Pbf, HMN, NNP);
        pcolT_conv<<<grids(HMN), 256, 0, stream>>>(S, cl, PtBf, MM, NNP, HH);
        // m0 = row-softmax(S) @ v1
        gemm(1, Pbf, NNP, (long long)MM * NNP, V1t, NNP, (long long)DD * NNP,
             ctxbf, CC, DD, nullptr, 0, nullptr, 0, 0,
             MM, DD, NNP, HH, 1.0f);
        gemm(0, ctxbf, CC, 0, WoT, CC, 0, cat + CC, 2 * CC, 0,
             bo, 0, nullptr, 0, 0, MM, CC, CC, 1, 1.0f);
        ffn(x0, x0bf, c1T + (size_t)l * 2 * CC * 2 * CC, cb1 + (size_t)l * 2 * CC,
            cg + (size_t)l * 2 * CC, cbeta + (size_t)l * 2 * CC,
            c2T + (size_t)l * CC * 2 * CC, cb2 + (size_t)l * CC);
        // m1 = col-softmax(S)^T @ v0
        gemm(1, PtBf, MM, (long long)NNP * MM, V0t, MM, (long long)DD * MM,
             ctxbf, CC, DD, nullptr, 0, nullptr, 0, 0,
             NNP, DD, MM, HH, 1.0f);
        gemm(0, ctxbf, CC, 0, WoT, CC, 0, cat + CC, 2 * CC, 0,
             bo, 0, nullptr, 0, 0, NNP, CC, CC, 1, 1.0f);
        ffn(x1, x1bf, c1T + (size_t)l * 2 * CC * 2 * CC, cb1 + (size_t)l * 2 * CC,
            cg + (size_t)l * 2 * CC, cbeta + (size_t)l * 2 * CC,
            c2T + (size_t)l * CC * 2 * CC, cb2 + (size_t)l * CC);
    };

    // ---- transformer layers ----
    for (int l = 0; l < LLAYERS; ++l) {
        selfblock(x0, x0bf, e0c, e0s, l);
        selfblock(x1, x1bf, e1c, e1s, l);
        crossblock(l);
    }

    // ---- assignment head ----
    gemm(1, x0bf, CC, 0, mWT, CC, 0, md0b, CC, 0, mb, 0, nullptr, 0, 0,
         MM, CC, CC, 1, 1.0f);
    gemm(1, x1bf, CC, 0, mWT, CC, 0, md1b, CC, 0, mb, 0, nullptr, 0, 0,
         NNP, CC, CC, 1, 1.0f);
    gemm(0, md0b, CC, 0, md1b, CC, 0, S, NNP, 0, nullptr, 0, nullptr, 0, 0,
         MM, NNP, CC, 1, 0.0625f);
    zproj<<<grids(MM), 256, 0, stream>>>(x0, zW, zb, z0, MM, CC);
    zproj<<<grids(NNP), 256, 0, stream>>>(x1, zW, zb, z1, NNP, CC);
    row_lse<<<MM, 256, 0, stream>>>(S, NNP, rl);
    col_lse<<<NNP, 256, 0, stream>>>(S, MM, NNP, cl);

    float* scores = (float*)d_out;
    size_t nsc = (size_t)(MM + 1) * (NNP + 1);
    assemble_scores<<<grids((long long)nsc), 256, 0, stream>>>(S, rl, cl, z0, z1, scores, MM, NNP);

    float* out_m0  = scores + nsc;
    float* out_m1  = out_m0 + MM;
    float* out_ms0 = out_m1 + NNP;
    float* out_ms1 = out_ms0 + MM;

    row_argmax<<<MM, 256, 0, stream>>>(scores, NNP + 1, NNP, max0, m0r);
    col_argmax<<<NNP, 256, 0, stream>>>(scores, NNP + 1, MM, m1r);
    filt0<<<grids(MM), 256, 0, stream>>>(m0r, m1r, max0, out_m0, out_ms0, MM, 0.1f);
    filt1<<<grids(NNP), 256, 0, stream>>>(m0r, m1r, out_ms0, out_m0, out_m1, out_ms1, NNP);
}
